// Attention_80479097192615
// MI455X (gfx1250) — matrix operand order repaired and checked by reading, not compiled, not run on hardware
//
#include <hip/hip_runtime.h>
#include <hip/hip_bf16.h>

#define IMG 128
#define NPATCH 32
#define DH 1024
#define CH 3
#define BATCH 256

typedef _Float16     h16   __attribute__((ext_vector_type(16)));
typedef float        f32x8 __attribute__((ext_vector_type(8)));
typedef unsigned int u32x8 __attribute__((ext_vector_type(8)));

__device__ __forceinline__ unsigned short h2us(float f) {
    _Float16 h = (_Float16)f;
    return __builtin_bit_cast(unsigned short, h);
}

// A-fragment (16x32 f16, M x K), source row-major [row][k] with stride 128 halves.
// lane L: row = rowBase + L%16 ; half j -> K = kbase + (j/8)*16 + 8*(L/16) + (j%8)
__device__ __forceinline__ h16 loadA(const unsigned short* lds, int row, int kbase, int g) {
    u32x8 u;
    const int rb = row * 128 + kbase;
#pragma unroll
    for (int p = 0; p < 8; ++p) {
        const int kb = (p < 4) ? (2 * p + 8 * g) : (2 * (p - 4) + 8 * g + 16);
        u[p] = *(const unsigned int*)(lds + rb + kb);
    }
    return __builtin_bit_cast(h16, u);
}

// B-fragment (32x16 f16, K x N) where B[k][n] = SRC[n][k], SRC row-major stride 128 halves.
// lane L: n = colBase + L%16 ; half j -> K = kbase + (j/8)*16 + 8*(L/16) + (j%8)
__device__ __forceinline__ h16 loadB(const unsigned short* lds, int col, int kbase, int g) {
    u32x8 u;
    const int rb = col * 128 + kbase;
#pragma unroll
    for (int p = 0; p < 8; ++p) {
        const int kb = (p < 4) ? (2 * p + 8 * g) : (2 * (p - 4) + 8 * g + 16);
        u[p] = *(const unsigned int*)(lds + rb + kb);
    }
    return __builtin_bit_cast(h16, u);
}

__device__ __forceinline__ f32x8 wmma(h16 a, h16 b, f32x8 c) {
    return __builtin_amdgcn_wmma_f32_16x16x32_f16(false, a, false, b, (short)0, c, false, false);
}

// Async global->LDS copy of one 64KB f32 image (16 bytes/lane/iter, ASYNCcnt-tracked).
__device__ __forceinline__ void async_img_load(unsigned ldsBase, const float* src, int tid) {
    const unsigned long long gbase = (unsigned long long)(uintptr_t)src;
#pragma unroll
    for (int i = 0; i < 16; ++i) {
        const unsigned byteOff = (unsigned)(tid + i * 256) * 16u;
        const unsigned laddr   = ldsBase + byteOff;
        asm volatile("global_load_async_to_lds_b128 %0, %1, %2 offset:0"
                     :: "v"(laddr), "v"(byteOff), "s"(gbase)
                     : "memory");
    }
}

__device__ __forceinline__ void wait_async0() {
    asm volatile("s_wait_asynccnt 0x0" ::: "memory");
}

__global__ __launch_bounds__(256) void draw_read_attention(
    const float* __restrict__ x, const float* __restrict__ x_hat,
    const float* __restrict__ h_dec, const float* __restrict__ W_read,
    const float* __restrict__ b_read, float* __restrict__ out)
{
    __shared__ __align__(16) float imgF32[IMG * IMG];     // 64 KB async staging (f32)
    __shared__ __align__(16) unsigned short imgH[IMG * IMG]; // 32 KB f16 image
    __shared__ unsigned short FXh[NPATCH * IMG];          // 8 KB
    __shared__ unsigned short FYh[NPATCH * IMG];          // 8 KB
    __shared__ unsigned short Tt[NPATCH * IMG];           // 8 KB, T transposed: Tt[m][h]
    __shared__ float red[256];
    __shared__ float rowsum[NPATCH * 8];
    __shared__ float pr[5];
    __shared__ float sp[8];                               // 0:gx 1:gy 2:sigsq 3:delta 4:gamma 5:inv2s
    __shared__ float muX[NPATCH], muY[NPATCH];

    const int tid  = threadIdx.x;
    const int lane = tid & 31;
    const int wv   = tid >> 5;
    const int lo   = lane & 15;
    const int g    = lane >> 4;

    const int blk = blockIdx.x;
    const int b   = blk / CH;
    const int c   = blk % CH;

    const size_t imgOff = (size_t)(b * CH + c) * (IMG * IMG);
    const unsigned stageLds = (unsigned)(uintptr_t)(&imgF32[0]);

    // Kick off the x-image fetch NOW so it overlaps the VALU preamble below.
    async_img_load(stageLds, x + imgOff, tid);

    // ---- params = h_dec[b] @ W_read^T + b_read  (5 block-wide reductions) ----
    const float* hb = h_dec + (size_t)b * DH;
#pragma unroll
    for (int p = 0; p < 5; ++p) {
        float acc = 0.f;
        const float* wp = W_read + p * DH;
        for (int i = tid; i < DH; i += 256) acc += hb[i] * wp[i];
        red[tid] = acc;
        __syncthreads();
        for (int s = 128; s > 0; s >>= 1) {
            if (tid < s) red[tid] += red[tid + s];
            __syncthreads();
        }
        if (tid == 0) pr[p] = red[0] + b_read[p];
        __syncthreads();
    }

    if (tid == 0) {
        const float gx     = (IMG + 1) * 0.5f * (pr[0] + 1.0f) - 0.5f;
        const float gy     = (IMG + 1) * 0.5f * (pr[1] + 1.0f) - 0.5f;
        const float sigsq  = expf(pr[2]);
        const float delta  = ((float)(IMG - 1) / (float)(NPATCH - 1)) * expf(pr[3]);
        sp[0] = gx; sp[1] = gy; sp[2] = sigsq; sp[3] = delta;
        sp[4] = expf(pr[4]);
        sp[5] = 1.0f / (2.0f * sigsq);
    }
    __syncthreads();
    if (tid < NPATCH)                 muX[tid]      = sp[0] + ((float)tid - 15.5f) * sp[3];
    else if (tid < 2 * NPATCH)        muY[tid - 32] = sp[1] + ((float)(tid - 32) - 15.5f) * sp[3];
    __syncthreads();

    // ---- filterbanks FX, FY -> f16 in LDS (8 threads per row of 128) ----
    {
        const int n = tid >> 3, q = tid & 7;
        const float inv2s = sp[5];
        float e[16];
        // FX (over h grid)
        float s = 0.f;
#pragma unroll
        for (int j = 0; j < 16; ++j) {
            const float d = (float)(q * 16 + j) - muX[n];
            e[j] = expf(-d * d * inv2s);
            s += e[j];
        }
        rowsum[n * 8 + q] = s;
        __syncthreads();
        float Z = 0.f;
#pragma unroll
        for (int k = 0; k < 8; ++k) Z += rowsum[n * 8 + k];
        const float invZ = 1.0f / (Z + 1e-8f);
#pragma unroll
        for (int j = 0; j < 16; ++j) FXh[n * 128 + q * 16 + j] = h2us(e[j] * invZ);
        __syncthreads();
        // FY (over w grid)
        s = 0.f;
#pragma unroll
        for (int j = 0; j < 16; ++j) {
            const float d = (float)(q * 16 + j) - muY[n];
            e[j] = expf(-d * d * inv2s);
            s += e[j];
        }
        rowsum[n * 8 + q] = s;
        __syncthreads();
        Z = 0.f;
#pragma unroll
        for (int k = 0; k < 8; ++k) Z += rowsum[n * 8 + k];
        const float invZ2 = 1.0f / (Z + 1e-8f);
#pragma unroll
        for (int j = 0; j < 16; ++j) FYh[n * 128 + q * 16 + j] = h2us(e[j] * invZ2);
    }

    for (int img = 0; img < 2; ++img) {
        // Staging buffer ready? (each wave waits for its own async batch, then all sync)
        wait_async0();
        __syncthreads();

        // Convert staged f32 image -> f16 imgH (v_cvt_pk_f16_f32 pairs).
        {
            const float4* s4 = (const float4*)imgF32;
            unsigned int* d32 = (unsigned int*)imgH;
#pragma unroll
            for (int i = 0; i < 16; ++i) {
                const int idx = tid + i * 256;
                const float4 v = s4[idx];
                d32[idx * 2 + 0] = (unsigned int)h2us(v.x) | ((unsigned int)h2us(v.y) << 16);
                d32[idx * 2 + 1] = (unsigned int)h2us(v.z) | ((unsigned int)h2us(v.w) << 16);
            }
        }
        __syncthreads();   // all staging reads done -> safe to refill imgF32

        if (img == 0) {
            // Prefetch x_hat asynchronously; overlaps the WMMA stages below.
            async_img_load(stageLds, x_hat + imgOff, tid);
        }

        // ---- Stage 1: T = img @ FY^T  (M=128 h, N=32 m, K=128 w) ----
        {
            const int ht = wv;                  // each wave: one 16-row tile, both N tiles
            f32x8 acc0 = {}, acc1 = {};
#pragma unroll
            for (int kt = 0; kt < 4; ++kt) {
                const h16 a  = loadA(imgH, ht * 16 + lo, kt * 32, g);
                const h16 b0 = loadB(FYh, lo,      kt * 32, g);
                const h16 b1 = loadB(FYh, 16 + lo, kt * 32, g);
                acc0 = wmma(a, b0, acc0);
                acc1 = wmma(a, b1, acc1);
            }
            // D layout: value v -> T[ht*16 + 8*g + v][ntile*16 + lo]; store transposed Tt[m][h]
#pragma unroll
            for (int v = 0; v < 8; ++v) {
                Tt[lo * 128        + ht * 16 + 8 * g + v] = h2us(acc0[v]);
                Tt[(16 + lo) * 128 + ht * 16 + 8 * g + v] = h2us(acc1[v]);
            }
        }
        __syncthreads();

        // ---- Stage 2: patch = FX @ T  (M=32 n, N=32 m, K=128 h), waves 0..3 ----
        if (wv < 4) {
            const int nt = wv >> 1;
            const int mt = wv & 1;
            f32x8 acc = {};
#pragma unroll
            for (int kt = 0; kt < 4; ++kt) {
                const h16 a  = loadA(FXh, nt * 16 + lo, kt * 32, g);
                const h16 bb = loadB(Tt,  mt * 16 + lo, kt * 32, g);
                acc = wmma(a, bb, acc);
            }
            const float gam = sp[4];
            float* ob = out + (size_t)b * 6144 + (size_t)img * 3072 + (size_t)c * 1024;
#pragma unroll
            for (int v = 0; v < 8; ++v) {
                ob[(nt * 16 + 8 * g + v) * 32 + mt * 16 + lo] = gam * acc[v];
            }
        }
        __syncthreads();   // stage-2 Tt reads done before next iteration rewrites it
    }
}

extern "C" void kernel_launch(void* const* d_in, const int* in_sizes, int n_in,
                              void* d_out, int out_size, void* d_ws, size_t ws_size,
                              hipStream_t stream) {
    (void)in_sizes; (void)n_in; (void)out_size; (void)d_ws; (void)ws_size;
    const float* x      = (const float*)d_in[0];
    const float* x_hat  = (const float*)d_in[1];
    const float* h_dec  = (const float*)d_in[2];
    const float* W_read = (const float*)d_in[3];
    const float* b_read = (const float*)d_in[4];
    float* out = (float*)d_out;

    dim3 grid(BATCH * CH);   // one workgroup per (batch, channel)
    dim3 block(256);         // 8 waves (wave32)
    hipLaunchKernelGGL(draw_read_attention, grid, block, 0, stream,
                       x, x_hat, h_dec, W_read, b_read, out);
}
